// Block_11115375362959
// MI455X (gfx1250) — matrix operand order repaired and checked by reading, not compiled, not run on hardware
//
#include <hip/hip_runtime.h>
#include <hip/hip_bf16.h>

typedef __attribute__((ext_vector_type(16))) _Float16 v16h;
typedef __attribute__((ext_vector_type(8)))  _Float16 v8h;
typedef __attribute__((ext_vector_type(4)))  _Float16 v4h;
typedef __attribute__((ext_vector_type(8)))  float    v8f;

union V16 { v16h v; v8h h[2]; };

__device__ __forceinline__ v8f zero8() {
    v8f z;
#pragma unroll
    for (int i = 0; i < 8; ++i) z[i] = 0.0f;
    return z;
}

// CDNA5 async copy: global -> LDS, 16 bytes per lane, tracked by ASYNCcnt.
__device__ __forceinline__ void async_g2l_b128(unsigned lds_off,
                                               const _Float16* gptr) {
    asm volatile("global_load_async_to_lds_b128 %0, %1, off"
                 :
                 : "v"(lds_off), "v"((unsigned long long)gptr)
                 : "memory");
}
__device__ __forceinline__ void wait_async0() {
    asm volatile("s_wait_asynccnt 0x0" ::: "memory");
}

// ---------------------------------------------------------------------------
// Weight packing: fp32 -> f16
// ---------------------------------------------------------------------------
__global__ __launch_bounds__(256) void pack_qkv_kernel(
    const float* __restrict__ wq, const float* __restrict__ wk,
    const float* __restrict__ wv, _Float16* __restrict__ out) {
    // out is [K=1024][N=3072] row-major; n = which*1024 + h*64 + d, k = c
    int idx = blockIdx.x * 256 + threadIdx.x;          // 1024*3072 total
    int n = idx % 3072;
    int c = idx / 3072;
    int which = n >> 10;
    int nn = n & 1023;
    int hh = nn >> 6, dd = nn & 63;
    const float* w = (which == 0) ? wq : ((which == 1) ? wk : wv);
    out[idx] = (_Float16)w[((size_t)hh * 1024 + c) * 64 + dd];
}

__global__ __launch_bounds__(256) void pack_f16_kernel(
    const float* __restrict__ in, _Float16* __restrict__ out, int n) {
    int i = blockIdx.x * 256 + threadIdx.x;
    if (i < n) out[i] = (_Float16)in[i];
}

// ---------------------------------------------------------------------------
// LayerNorm (block per row of 1024) -> f16 output
// ---------------------------------------------------------------------------
__global__ __launch_bounds__(256) void ln_f16_kernel(
    const float* __restrict__ x, const float* __restrict__ g,
    const float* __restrict__ bt, _Float16* __restrict__ out) {
    __shared__ float s1[256];
    __shared__ float s2[256];
    const int row = blockIdx.x;
    const int tid = threadIdx.x;
    const float4 v = *(const float4*)&x[(size_t)row * 1024 + tid * 4];
    s1[tid] = v.x + v.y + v.z + v.w;
    s2[tid] = v.x * v.x + v.y * v.y + v.z * v.z + v.w * v.w;
    __syncthreads();
#pragma unroll
    for (int st = 128; st > 0; st >>= 1) {
        if (tid < st) { s1[tid] += s1[tid + st]; s2[tid] += s2[tid + st]; }
        __syncthreads();
    }
    const float mu  = s1[0] * (1.0f / 1024.0f);
    const float var = s2[0] * (1.0f / 1024.0f) - mu * mu;
    const float rs  = rsqrtf(var + 1e-5f);
    const int c = tid * 4;
    v4h o;
    o[0] = (_Float16)((v.x - mu) * rs * g[c + 0] + bt[c + 0]);
    o[1] = (_Float16)((v.y - mu) * rs * g[c + 1] + bt[c + 1]);
    o[2] = (_Float16)((v.z - mu) * rs * g[c + 2] + bt[c + 2]);
    o[3] = (_Float16)((v.w - mu) * rs * g[c + 3] + bt[c + 3]);
    *(v4h*)&out[(size_t)row * 1024 + c] = o;
}

// ---------------------------------------------------------------------------
// WMMA GEMM: C[M,N] = A[M,K] (f16, row-major) * B[K,N] (f16, K-major)
// BM=128 BN=64 BK=64, 8 waves (4 along M x 2 along N), 2x2 WMMA tiles per wave,
// double-buffered async (ASYNCcnt) global->LDS staging.
// ---------------------------------------------------------------------------
struct GemmP {
    const _Float16* A;
    const _Float16* B;
    int M, N, K;
    const float* bias;       // EPI 1,2
    const float* res;        // EPI 1
    float*       outF;       // EPI 1
    _Float16*    out16;      // EPI 2
    _Float16*    q;          // EPI 0
    _Float16*    kt;         // EPI 0
    _Float16*    v;          // EPI 0
};

#define BM 128
#define BN 64
#define BK 64

template <int EPI>
__global__ __launch_bounds__(256) void gemm_wmma_kernel(GemmP p) {
    __shared__ _Float16 As[2][BM][BK + 8];   // 2 x 128 x 72 f16 = 36 KB
    __shared__ _Float16 Bs[2][BK][BN + 8];   // 2 x  64 x 72 f16 = 18 KB

    const int tid  = threadIdx.x;
    const int lane = tid & 31;
    const int wave = tid >> 5;
    const int wm   = (wave >> 1) * 32;
    const int wn   = (wave & 1) * 32;
    const int bm   = blockIdx.y * BM;
    const int bn   = blockIdx.x * BN;
    const int r    = lane & 15;
    const int hi   = lane >> 4;

    v8f acc[2][2];
#pragma unroll
    for (int tm = 0; tm < 2; ++tm)
#pragma unroll
        for (int tn = 0; tn < 2; ++tn) acc[tm][tn] = zero8();

    // async stage mapping: 256 threads x 16B; A tile 16KB (4 issues),
    // B tile 8KB (2 issues). 32 rows x 64 cols per pass.
    const int lrow = tid >> 3;          // 0..31
    const int lcol = (tid & 7) * 8;     // 0..56

    auto issueTile = [&](int kt0, int buf) {
#pragma unroll
        for (int pa = 0; pa < 4; ++pa) {
            const int rr2 = pa * 32 + lrow;
            async_g2l_b128((unsigned)(size_t)&As[buf][rr2][lcol],
                           &p.A[(size_t)(bm + rr2) * p.K + kt0 + lcol]);
        }
#pragma unroll
        for (int pb = 0; pb < 2; ++pb) {
            const int rr2 = pb * 32 + lrow;
            async_g2l_b128((unsigned)(size_t)&Bs[buf][rr2][lcol],
                           &p.B[(size_t)(kt0 + rr2) * p.N + bn + lcol]);
        }
    };

    const int nk = p.K / BK;
    issueTile(0, 0);
    int buf = 0;
    for (int i = 0; i < nk; ++i) {
        wait_async0();        // own tile-i fills done
        __syncthreads();      // everyone's fills done; prev buffer free
        if (i + 1 < nk) issueTile((i + 1) * BK, buf ^ 1);  // overlap compute

#pragma unroll
        for (int kc = 0; kc < 2; ++kc) {
            // A fragment: lane<16 -> K {0..7,16..23}; lane>=16 -> {8..15,24..31}
            V16 aF[2], bF[2];
#pragma unroll
            for (int tm = 0; tm < 2; ++tm) {
                const int ar = wm + tm * 16 + r;
                aF[tm].h[0] = *(const v8h*)&As[buf][ar][kc * 32 + hi * 8];
                aF[tm].h[1] = *(const v8h*)&As[buf][ar][kc * 32 + 16 + hi * 8];
            }
            // B fragment: lane%16 = N column; K same as A: {0..7,16..23} / {8..15,24..31}
#pragma unroll
            for (int tn = 0; tn < 2; ++tn) {
                const int bc = wn + tn * 16;
#pragma unroll
                for (int s = 0; s < 16; ++s)
                    bF[tn].v[s] = Bs[buf][kc * 32 + (s >> 3) * 16 + hi * 8 + (s & 7)][bc + r];
            }
#pragma unroll
            for (int tm = 0; tm < 2; ++tm)
#pragma unroll
                for (int tn = 0; tn < 2; ++tn)
                    acc[tm][tn] = __builtin_amdgcn_wmma_f32_16x16x32_f16(
                        false, aF[tm].v, false, bF[tn].v, (short)0,
                        acc[tm][tn], false, false);
        }
        buf ^= 1;
    }

    // Epilogue. C layout: VGPR rr, lane<16 -> M=rr,N=lane; lane>=16 -> M=8+rr
#pragma unroll
    for (int tm = 0; tm < 2; ++tm)
#pragma unroll
        for (int tn = 0; tn < 2; ++tn)
#pragma unroll
            for (int rr = 0; rr < 8; ++rr) {
                const int row = bm + wm + tm * 16 + hi * 8 + rr;
                const int col = bn + wn + tn * 16 + r;
                const float val = acc[tm][tn][rr];
                if (EPI == 0) {
                    // QKV scatter. row = b*256+t ; col = which*1024 + h*64 + d
                    const int which = col >> 10;
                    const int nn = col & 1023;
                    const int hh = nn >> 6, dd = nn & 63;
                    const int bb = row >> 8, tt = row & 255;
                    const int bh = bb * 16 + hh;
                    if (which == 0) {
                        // fold 1/sqrt(64) into Q
                        p.q[((size_t)bh * 256 + tt) * 64 + dd] =
                            (_Float16)(val * 0.125f);
                    } else if (which == 1) {
                        // K stored transposed: [B,H,HD,T]
                        p.kt[((size_t)bh * 64 + dd) * 256 + tt] = (_Float16)val;
                    } else {
                        p.v[((size_t)bh * 256 + tt) * 64 + dd] = (_Float16)val;
                    }
                } else if (EPI == 1) {
                    p.outF[(size_t)row * p.N + col] =
                        val + p.bias[col] + p.res[(size_t)row * p.N + col];
                } else {
                    const float xg = val + p.bias[col];
                    const float ge =
                        0.5f * xg * (1.0f + erff(xg * 0.70710678118654752f));
                    p.out16[(size_t)row * p.N + col] = (_Float16)ge;
                }
            }
}

// ---------------------------------------------------------------------------
// Flash attention: one wave per 16-query block of one (b,h) head.
// Q pre-scaled; Kt is [B,H,HD,T]; V is [B,H,T,HD]; O is [B,T,H*HD] f16.
// ---------------------------------------------------------------------------
__global__ __launch_bounds__(128) void attn_wmma_kernel(
    const _Float16* __restrict__ Q, const _Float16* __restrict__ Kt,
    const _Float16* __restrict__ V, _Float16* __restrict__ O) {
    __shared__ _Float16 Pst[4][16][40];   // per-wave P staging (C->A layout)

    const int tid  = threadIdx.x;
    const int lane = tid & 31;
    const int wave = tid >> 5;
    const int gw   = blockIdx.x * 4 + wave;
    const int qb   = gw & 15;             // query block (16 rows)
    const int bh   = gw >> 4;             // b*16 + h
    const int b    = bh >> 4;
    const int h    = bh & 15;
    const int r    = lane & 15;
    const int hi   = lane >> 4;

    // Q fragments (16x64 = two 16x32 chunks), loaded straight from global
    const _Float16* qp = Q + ((size_t)bh * 256 + qb * 16) * 64;
    V16 qA[2];
#pragma unroll
    for (int kc = 0; kc < 2; ++kc) {
        qA[kc].h[0] = *(const v8h*)&qp[r * 64 + kc * 32 + hi * 8];
        qA[kc].h[1] = *(const v8h*)&qp[r * 64 + kc * 32 + 16 + hi * 8];
    }

    float mrow[8], lrow[8];
#pragma unroll
    for (int i = 0; i < 8; ++i) { mrow[i] = -INFINITY; lrow[i] = 0.0f; }
    v8f oacc[4];
#pragma unroll
    for (int i = 0; i < 4; ++i) oacc[i] = zero8();

    const int qend = qb * 16 + 15;
    for (int s0 = 0; s0 <= qend; s0 += 32) {
        // ---- scores: S = Q * K^T over HD=64 (two k-chunks), 2 key tiles ----
        V16 kF[2][2];
#pragma unroll
        for (int kc = 0; kc < 2; ++kc)
#pragma unroll
            for (int nt = 0; nt < 2; ++nt) {
                const _Float16* kp =
                    Kt + ((size_t)bh * 64 + kc * 32) * 256 + s0 + nt * 16 + r;
#pragma unroll
                for (int s = 0; s < 16; ++s)
                    kF[kc][nt].v[s] = kp[((s >> 3) * 16 + hi * 8 + (s & 7)) * 256];
            }
        v8f sacc[2];
        sacc[0] = zero8();
        sacc[1] = zero8();
#pragma unroll
        for (int kc = 0; kc < 2; ++kc)
#pragma unroll
            for (int nt = 0; nt < 2; ++nt)
                sacc[nt] = __builtin_amdgcn_wmma_f32_16x16x32_f16(
                    false, qA[kc].v, false, kF[kc][nt].v, (short)0, sacc[nt],
                    false, false);

        // ---- causal mask + online softmax (rows live in 16-lane halves) ----
#pragma unroll
        for (int rr = 0; rr < 8; ++rr) {
            const int qrow = qb * 16 + hi * 8 + rr;
#pragma unroll
            for (int nt = 0; nt < 2; ++nt) {
                const int colk = s0 + nt * 16 + r;
                if (colk > qrow) sacc[nt][rr] = -INFINITY;
            }
            float mx = fmaxf(sacc[0][rr], sacc[1][rr]);
#pragma unroll
            for (int off = 8; off >= 1; off >>= 1)
                mx = fmaxf(mx, __shfl_xor(mx, off, 32));
            const float mn = fmaxf(mrow[rr], mx);
            const float alpha = __expf(mrow[rr] - mn);
            const float p0 = __expf(sacc[0][rr] - mn);
            const float p1 = __expf(sacc[1][rr] - mn);
            sacc[0][rr] = p0;
            sacc[1][rr] = p1;
            float rs = p0 + p1;
#pragma unroll
            for (int off = 8; off >= 1; off >>= 1)
                rs += __shfl_xor(rs, off, 32);
            lrow[rr] = lrow[rr] * alpha + rs;
            mrow[rr] = mn;
#pragma unroll
            for (int nh = 0; nh < 4; ++nh) oacc[nh][rr] *= alpha;
        }

        // ---- P: C-layout -> A-layout via per-wave LDS staging ----
#pragma unroll
        for (int rr = 0; rr < 8; ++rr) {
            Pst[wave][hi * 8 + rr][r]      = (_Float16)sacc[0][rr];
            Pst[wave][hi * 8 + rr][16 + r] = (_Float16)sacc[1][rr];
        }
        asm volatile("s_wait_dscnt 0x0" ::: "memory");
        V16 pA;
        pA.h[0] = *(const v8h*)&Pst[wave][r][hi * 8];
        pA.h[1] = *(const v8h*)&Pst[wave][r][16 + hi * 8];

        // ---- O += P * V (K = 32 keys, 4 tiles over HD) ----
        const _Float16* vp = V + ((size_t)bh * 256 + s0) * 64 + r;
#pragma unroll
        for (int nh = 0; nh < 4; ++nh) {
            V16 vF;
#pragma unroll
            for (int s = 0; s < 16; ++s)
                vF.v[s] = vp[((s >> 3) * 16 + hi * 8 + (s & 7)) * 64 + nh * 16];
            oacc[nh] = __builtin_amdgcn_wmma_f32_16x16x32_f16(
                false, pA.v, false, vF.v, (short)0, oacc[nh], false, false);
        }
    }

    // ---- finalize: O / l, scatter into concat-head layout [B,T,1024] ----
#pragma unroll
    for (int nh = 0; nh < 4; ++nh)
#pragma unroll
        for (int rr = 0; rr < 8; ++rr) {
            const int tt = qb * 16 + hi * 8 + rr;
            const int dd = nh * 16 + r;
            const float ov = oacc[nh][rr] / lrow[rr];
            O[((size_t)b * 256 + tt) * 1024 + h * 64 + dd] = (_Float16)ov;
        }
}

// ---------------------------------------------------------------------------
// Host-side launch
// ---------------------------------------------------------------------------
extern "C" void kernel_launch(void* const* d_in, const int* in_sizes, int n_in,
                              void* d_out, int out_size, void* d_ws,
                              size_t ws_size, hipStream_t stream) {
    (void)in_sizes; (void)n_in; (void)out_size; (void)ws_size;
    const float* x    = (const float*)d_in[0];
    const float* ln1g = (const float*)d_in[1];
    const float* ln1b = (const float*)d_in[2];
    const float* wq   = (const float*)d_in[3];
    const float* wk   = (const float*)d_in[4];
    const float* wv   = (const float*)d_in[5];
    const float* wpj  = (const float*)d_in[6];
    const float* bpj  = (const float*)d_in[7];
    const float* ln2g = (const float*)d_in[8];
    const float* ln2b = (const float*)d_in[9];
    const float* w1   = (const float*)d_in[10];
    const float* b1   = (const float*)d_in[11];
    const float* w2   = (const float*)d_in[12];
    const float* b2   = (const float*)d_in[13];
    float* out = (float*)d_out;

    const size_t MT = 16384;  // B*T
    char* ws = (char*)d_ws;
    size_t off = 0;
    auto alloc = [&](size_t bytes) -> void* {
        void* p = ws + off;
        off = (off + bytes + 255) & ~(size_t)255;
        return p;
    };
    _Float16* h16     = (_Float16*)alloc(MT * 1024 * 2);
    _Float16* wqkv16  = (_Float16*)alloc((size_t)1024 * 3072 * 2);
    _Float16* wproj16 = (_Float16*)alloc((size_t)1024 * 1024 * 2);
    _Float16* w1_16   = (_Float16*)alloc((size_t)1024 * 4096 * 2);
    _Float16* w2_16   = (_Float16*)alloc((size_t)4096 * 1024 * 2);
    _Float16* q16     = (_Float16*)alloc(MT * 1024 * 2);
    _Float16* kt16    = (_Float16*)alloc(MT * 1024 * 2);
    _Float16* v16     = (_Float16*)alloc(MT * 1024 * 2);
    _Float16* o16     = (_Float16*)alloc(MT * 1024 * 2);
    float*    x2      = (float*)   alloc(MT * 1024 * 4);
    _Float16* h2_16   = (_Float16*)alloc(MT * 1024 * 2);
    _Float16* hff16   = (_Float16*)alloc(MT * 4096 * 2);

    // weight packing (fp32 -> f16, QKV concatenated K-major)
    pack_qkv_kernel<<<(1024 * 3072) / 256, 256, 0, stream>>>(wq, wk, wv, wqkv16);
    pack_f16_kernel<<<(1024 * 1024) / 256, 256, 0, stream>>>(wpj, wproj16, 1024 * 1024);
    pack_f16_kernel<<<(1024 * 4096) / 256, 256, 0, stream>>>(w1, w1_16, 1024 * 4096);
    pack_f16_kernel<<<(4096 * 1024) / 256, 256, 0, stream>>>(w2, w2_16, 4096 * 1024);

    // LN1 -> h16
    ln_f16_kernel<<<(int)MT, 256, 0, stream>>>(x, ln1g, ln1b, h16);

    // QKV GEMM: [16384,1024] x [1024,3072] with scatter epilogue
    {
        GemmP p{};
        p.A = h16; p.B = wqkv16; p.M = (int)MT; p.N = 3072; p.K = 1024;
        p.q = q16; p.kt = kt16; p.v = v16;
        gemm_wmma_kernel<0><<<dim3(3072 / BN, MT / BM), 256, 0, stream>>>(p);
    }

    // attention: 1024 heads * 16 q-blocks, 4 waves/block
    attn_wmma_kernel<<<(64 * 16 * 16) / 4, 128, 0, stream>>>(q16, kt16, v16, o16);

    // output projection + bias + residual(x) -> x2 (f32)
    {
        GemmP p{};
        p.A = o16; p.B = wproj16; p.M = (int)MT; p.N = 1024; p.K = 1024;
        p.bias = bpj; p.res = x; p.outF = x2;
        gemm_wmma_kernel<1><<<dim3(1024 / BN, MT / BM), 256, 0, stream>>>(p);
    }

    // LN2 -> h2_16
    ln_f16_kernel<<<(int)MT, 256, 0, stream>>>(x2, ln2g, ln2b, h2_16);

    // FF1 + bias + exact GELU -> hff16
    {
        GemmP p{};
        p.A = h2_16; p.B = w1_16; p.M = (int)MT; p.N = 4096; p.K = 1024;
        p.bias = b1; p.out16 = hff16;
        gemm_wmma_kernel<2><<<dim3(4096 / BN, MT / BM), 256, 0, stream>>>(p);
    }

    // FF2 + bias + residual(x2) -> out (f32)
    {
        GemmP p{};
        p.A = hff16; p.B = w2_16; p.M = (int)MT; p.N = 1024; p.K = 4096;
        p.bias = b2; p.res = x2; p.outF = out;
        gemm_wmma_kernel<1><<<dim3(1024 / BN, MT / BM), 256, 0, stream>>>(p);
    }
}
